// SNN_16234976379184
// MI455X (gfx1250) — matrix operand order repaired and checked by reading, not compiled, not run on hardware
//
#include <hip/hip_runtime.h>

typedef __attribute__((ext_vector_type(16))) _Float16 v16h;
typedef __attribute__((ext_vector_type(8)))  _Float16 v8h;
typedef __attribute__((ext_vector_type(8)))  float    v8f;

#define B_   256
#define IN_  1024
#define H_   2048
#define OUT_ 1024
#define T_   50
#define HW_  (H_ / 32)        // 64 spike words per (b) row
#define PLANE_ (B_ * HW_)     // 16384 words per time plane

// ---------------------------------------------------------------------------
// f32 -> f16 conversion (weights, x, spike counts)
// ---------------------------------------------------------------------------
__global__ __launch_bounds__(256)
void f32_to_f16_kernel(const float* __restrict__ src, _Float16* __restrict__ dst, int n) {
    int i = blockIdx.x * blockDim.x + threadIdx.x;
    if (i < n) dst[i] = (_Float16)src[i];
}

__global__ __launch_bounds__(256)
void zero_kernel(float* __restrict__ p, int n) {
    int i = blockIdx.x * blockDim.x + threadIdx.x;
    if (i < n) p[i] = 0.0f;
}

// ---------------------------------------------------------------------------
// GEMM: C[M x N] = alpha * (A[M x K] @ Bm[N x K]^T) + bias[N]
// f16 inputs, f32 WMMA accumulate. Wave tile 16x64 (4 WMMAs share A frag),
// block = 8 waves -> 32x256 tile. Layouts per CDNA5 ISA 7.12.2.
// ---------------------------------------------------------------------------
__global__ __launch_bounds__(256)
void gemm_f16_kernel(const _Float16* __restrict__ A, int lda,
                     const _Float16* __restrict__ Bm, int ldb,
                     const float* __restrict__ bias,
                     float* __restrict__ C, int ldc,
                     int K, float alpha)
{
    const int lane = threadIdx.x & 31;
    const int wave = threadIdx.x >> 5;
    const int half = lane >> 4;     // lane group (0: lanes 0-15, 1: lanes 16-31)
    const int l16  = lane & 15;

    const int tileM = blockIdx.y * 32  + (wave >> 2) * 16;
    const int tileN = blockIdx.x * 256 + (wave & 3) * 64;

    v8f acc[4] = {};

    // A: per-lane row = tileM + l16; element e -> k = e + (e>=8?8:0) + half*8
    const _Float16* aRow = A + (size_t)(tileM + l16) * lda + half * 8;
    // B: per-lane col = tileN + l16 (+16*nt); element e -> k = e + (e>=8?8:0) + half*8
    const _Float16* bCol = Bm + (size_t)(tileN + l16) * ldb + half * 8;

    for (int k = 0; k < K; k += 32) {
        v8h lo = *(const v8h*)(aRow + k);        // k+half*8 + [0..7]
        v8h hi = *(const v8h*)(aRow + k + 16);   // k+16+half*8 + [0..7]
        v16h a;
#pragma unroll
        for (int e = 0; e < 8; ++e) { a[e] = lo[e]; a[8 + e] = hi[e]; }

#pragma unroll
        for (int nt = 0; nt < 4; ++nt) {
            const _Float16* bp = bCol + (size_t)(16 * nt) * ldb + k;
            v8h blo = *(const v8h*)(bp);
            v8h bhi = *(const v8h*)(bp + 16);
            v16h b;
#pragma unroll
            for (int e = 0; e < 8; ++e) { b[e] = blo[e]; b[8 + e] = bhi[e]; }
            acc[nt] = __builtin_amdgcn_wmma_f32_16x16x32_f16(
                false, a, false, b, (short)0, acc[nt], false, false);
        }
    }

    // D layout: row = r + 8*half, col = l16
#pragma unroll
    for (int r = 0; r < 8; ++r) {
        int row = tileM + r + half * 8;
#pragma unroll
        for (int nt = 0; nt < 4; ++nt) {
            int col = tileN + nt * 16 + l16;
            C[(size_t)row * ldc + col] = alpha * acc[nt][r] + bias[col];
        }
    }
}

// ---------------------------------------------------------------------------
// Layer-1 LIF recurrence: i1 fixed per step; emit T bit-planes of spikes.
// One thread owns 32 consecutive h of one batch row (one spike word).
// ---------------------------------------------------------------------------
__global__ __launch_bounds__(256)
void lif1_kernel(const float* __restrict__ i1, unsigned* __restrict__ s1p)
{
    int idx = blockIdx.x * blockDim.x + threadIdx.x;   // (b, hword)
    int b  = idx / HW_;
    int hw = idx % HW_;

    float cur[32], v[32];
    const float* ip = i1 + (size_t)b * H_ + hw * 32;
#pragma unroll
    for (int j = 0; j < 32; ++j) { cur[j] = ip[j]; v[j] = 0.0f; }

    unsigned* out = s1p + (size_t)b * HW_ + hw;
    for (int t = 0; t < T_; ++t) {
        unsigned w = 0;
#pragma unroll
        for (int j = 0; j < 32; ++j) {
            v[j] += cur[j];
            if (v[j] >= 1.0f) { w |= (1u << j); v[j] = 0.0f; }
        }
        out[(size_t)t * PLANE_] = w;
    }
}

// ---------------------------------------------------------------------------
// One timestep of layer 2, fused:
//   i2 = expand_bits(s1(t)) @ W2^T           (WMMA, A built from spike bits)
//   v2 += i2 + b2; s2 = v2>=1; v2 *= (1-s2); cnt2 += s2
// ---------------------------------------------------------------------------
__global__ __launch_bounds__(256)
void lif2_step_kernel(const unsigned* __restrict__ s1plane,
                      const _Float16* __restrict__ W2h,
                      const float* __restrict__ b2,
                      float* __restrict__ v2,
                      float* __restrict__ cnt2)
{
    const int lane = threadIdx.x & 31;
    const int wave = threadIdx.x >> 5;
    const int half = lane >> 4;
    const int l16  = lane & 15;

    const int tileM = blockIdx.y * 32  + (wave >> 2) * 16;
    const int tileN = blockIdx.x * 256 + (wave & 3) * 64;

    v8f acc[4] = {};

    const unsigned* sp = s1plane + (size_t)(tileM + l16) * HW_;
    const _Float16* bCol = W2h + (size_t)(tileN + l16) * H_ + half * 8;

    const _Float16 ONE = (_Float16)1.0f, ZERO = (_Float16)0.0f;

    for (int k = 0; k < H_; k += 32) {
        unsigned w = sp[k >> 5];
        v16h a;
#pragma unroll
        for (int e = 0; e < 8; ++e) {
            a[e]     = ((w >> (half * 8 + e))      & 1u) ? ONE : ZERO;  // k = e + half*8
            a[8 + e] = ((w >> (16 + half * 8 + e)) & 1u) ? ONE : ZERO;  // k = e + 16 + half*8
        }
#pragma unroll
        for (int nt = 0; nt < 4; ++nt) {
            const _Float16* bp = bCol + (size_t)(16 * nt) * H_ + k;
            v8h blo = *(const v8h*)(bp);
            v8h bhi = *(const v8h*)(bp + 16);
            v16h b;
#pragma unroll
            for (int e = 0; e < 8; ++e) { b[e] = blo[e]; b[8 + e] = bhi[e]; }
            acc[nt] = __builtin_amdgcn_wmma_f32_16x16x32_f16(
                false, a, false, b, (short)0, acc[nt], false, false);
        }
    }

    // membrane update epilogue (each output element owned by exactly one lane)
#pragma unroll
    for (int r = 0; r < 8; ++r) {
        int row = tileM + r + half * 8;
#pragma unroll
        for (int nt = 0; nt < 4; ++nt) {
            int col = tileN + nt * 16 + l16;
            size_t o = (size_t)row * H_ + col;
            float vv = v2[o] + acc[nt][r] + b2[col];
            float s  = (vv >= 1.0f) ? 1.0f : 0.0f;
            v2[o]   = vv * (1.0f - s);
            cnt2[o] += s;
        }
    }
}

// ---------------------------------------------------------------------------
// Launch:  convert -> gemm1 -> lif1 bits -> 50x fused lif2 GEMM -> gemm3
// out = (cnt2 @ W3^T) / T + b3
// ---------------------------------------------------------------------------
extern "C" void kernel_launch(void* const* d_in, const int* in_sizes, int n_in,
                              void* d_out, int out_size, void* d_ws, size_t ws_size,
                              hipStream_t stream)
{
    (void)in_sizes; (void)n_in; (void)out_size; (void)ws_size;

    const float* x  = (const float*)d_in[0];
    const float* W1 = (const float*)d_in[1];
    const float* b1 = (const float*)d_in[2];
    const float* W2 = (const float*)d_in[3];
    const float* b2 = (const float*)d_in[4];
    const float* W3 = (const float*)d_in[5];
    const float* b3 = (const float*)d_in[6];
    float* out = (float*)d_out;

    char* ws = (char*)d_ws;
    size_t off = 0;
    auto alloc = [&](size_t bytes) -> char* {
        char* p = ws + off;
        off += (bytes + 255) & ~(size_t)255;
        return p;
    };
    _Float16* W1h = (_Float16*)alloc((size_t)H_ * IN_ * 2);   // 4 MB
    _Float16* W2h = (_Float16*)alloc((size_t)H_ * H_ * 2);    // 8 MB (L2-resident)
    _Float16* W3h = (_Float16*)alloc((size_t)OUT_ * H_ * 2);  // 4 MB
    _Float16* xh  = (_Float16*)alloc((size_t)B_ * IN_ * 2);
    float*    i1  = (float*)alloc((size_t)B_ * H_ * 4);
    unsigned* s1p = (unsigned*)alloc((size_t)T_ * PLANE_ * 4);// 3.3 MB spike bit-planes
    float*    v2  = (float*)alloc((size_t)B_ * H_ * 4);
    float*    cnt2= (float*)alloc((size_t)B_ * H_ * 4);
    _Float16* c2h = (_Float16*)alloc((size_t)B_ * H_ * 2);

    dim3 blk(256);

    f32_to_f16_kernel<<<(H_ * IN_ + 255) / 256, blk, 0, stream>>>(W1, W1h, H_ * IN_);
    f32_to_f16_kernel<<<(H_ * H_  + 255) / 256, blk, 0, stream>>>(W2, W2h, H_ * H_);
    f32_to_f16_kernel<<<(OUT_ * H_ + 255) / 256, blk, 0, stream>>>(W3, W3h, OUT_ * H_);
    f32_to_f16_kernel<<<(B_ * IN_ + 255) / 256, blk, 0, stream>>>(x, xh, B_ * IN_);
    zero_kernel<<<(B_ * H_ + 255) / 256, blk, 0, stream>>>(v2,   B_ * H_);
    zero_kernel<<<(B_ * H_ + 255) / 256, blk, 0, stream>>>(cnt2, B_ * H_);

    // i1 = x @ W1^T + b1   (M=256, N=2048, K=1024)
    gemm_f16_kernel<<<dim3(H_ / 256, B_ / 32), blk, 0, stream>>>(
        xh, IN_, W1h, IN_, b1, i1, H_, IN_, 1.0f);

    // precompute all layer-1 spike bit-planes
    lif1_kernel<<<(B_ * HW_) / 256, blk, 0, stream>>>(i1, s1p);

    // sequential membrane recurrence for layer 2 (each step = fused WMMA GEMM)
    for (int t = 0; t < T_; ++t)
        lif2_step_kernel<<<dim3(H_ / 256, B_ / 32), blk, 0, stream>>>(
            s1p + (size_t)t * PLANE_, W2h, b2, v2, cnt2);

    // layer 3 collapses: out = (sum_t s2(t)) @ W3^T / T + b3
    f32_to_f16_kernel<<<(B_ * H_ + 255) / 256, blk, 0, stream>>>(cnt2, c2h, B_ * H_);
    gemm_f16_kernel<<<dim3(OUT_ / 256, B_ / 32), blk, 0, stream>>>(
        c2h, H_, W3h, H_, b3, out, OUT_, H_, 1.0f / (float)T_);
}
